// Policy_747324310113
// MI455X (gfx1250) — matrix operand order repaired and checked by reading, not compiled, not run on hardware
//
#include <hip/hip_runtime.h>
#include <hip/hip_bf16.h>
#include <math.h>

typedef __attribute__((ext_vector_type(16))) _Float16 v16h;
typedef __attribute__((ext_vector_type(8)))  float    v8f;

#define GRIDX 160          // blocks per dataset for the WMMA pass
#define NEG_INF (-3.402823466e38f)

__device__ __forceinline__ float vmax8(const v8f& d) {
  return fmaxf(fmaxf(fmaxf(d[0], d[1]), fmaxf(d[2], d[3])),
               fmaxf(fmaxf(d[4], d[5]), fmaxf(d[6], d[7])));
}

// ---------------------------------------------------------------------------
// Pass 1: per-unit MLP (layer1 f32 VALU, layer2 f16 WMMA w/ f32 accum) + max
// emb[m][n] = sum_k relu(u[m]·W1[k]+b1[k]) * W2[n][k]   (b2 added later)
// Each wave: 16 units x 128 cols per tile; B (W2) held in VGPRs for reuse.
// All 8 WMMAs issue into 8 live accumulators; a sched_barrier keeps the
// consuming v_max chains after the whole WMMA group so the WMMA->VALU RAW
// hazard window is filled with useful instructions instead of v_nop.
// ---------------------------------------------------------------------------
__global__ __launch_bounds__(256) void k_unit_mlp_max(
    const float* __restrict__ U, const float* __restrict__ W1,
    const float* __restrict__ B1, const float* __restrict__ W2,
    float* __restrict__ pmax, int N)
{
  const int tid   = threadIdx.x;
  const int lane  = tid & 31;
  const int wave  = tid >> 5;
  const int laneM = lane & 15;
  const int hi    = (lane >= 16) ? 1 : 0;

  // B tiles: v16h element e of lane L holds W2[n][k], n = 16*ct + (L&15),
  // k = 16*hi + e   (contiguous k order, same as A; wave32)
  v16h B[8];
  {
    const int kb = hi * 16;
    #pragma unroll
    for (int ct = 0; ct < 8; ++ct) {
      const float* w = W2 + (ct * 16 + laneM) * 32 + kb;
      #pragma unroll
      for (int e = 0; e < 16; ++e) B[ct][e] = (_Float16)w[e];
    }
  }

  // Register-cache layer-1 rows this lane needs for the A layout:
  // A element e <-> k = 16*hi + e
  float w1r[16][3]; float b1r[16];
  #pragma unroll
  for (int e = 0; e < 16; ++e) {
    int k = hi * 16 + e;
    w1r[e][0] = W1[k*3+0]; w1r[e][1] = W1[k*3+1]; w1r[e][2] = W1[k*3+2];
    b1r[e] = B1[k];
  }

  float runmax[8];
  #pragma unroll
  for (int ct = 0; ct < 8; ++ct) runmax[ct] = NEG_INF;

  const v8f zero = {};
  const int tiles = N >> 4;
  for (int t = blockIdx.x * 8 + wave; t < tiles; t += GRIDX * 8) {
    const int m = t * 16 + laneM;
    const float u0 = U[m*3+0], u1 = U[m*3+1], u2 = U[m*3+2];
    v16h A;
    #pragma unroll
    for (int e = 0; e < 16; ++e) {
      float h = fmaxf(u0*w1r[e][0] + u1*w1r[e][1] + u2*w1r[e][2] + b1r[e], 0.0f);
      A[e] = (_Float16)h;
    }

    // Issue all 8 WMMAs into distinct live accumulators.
    v8f c[8];
    #pragma unroll
    for (int ct = 0; ct < 8; ++ct)
      c[ct] = __builtin_amdgcn_wmma_f32_16x16x32_f16(false, A, false, B[ct],
                                                     (short)0, zero,
                                                     false, false);
#if defined(__has_builtin)
#if __has_builtin(__builtin_amdgcn_sched_barrier)
    // 0x70: only VMEM may be scheduled across (lets next-tile loads hoist),
    // VALU/WMMA must stay on their side -> max chains cannot sink under
    // their producing WMMA.
    __builtin_amdgcn_sched_barrier(0x70);
#endif
#endif
    #pragma unroll
    for (int ct = 0; ct < 8; ++ct)
      runmax[ct] = fmaxf(runmax[ct], vmax8(c[ct]));
  }

  // combine M-halves (lanes L / L+16 hold same column, different M range)
  #pragma unroll
  for (int ct = 0; ct < 8; ++ct)
    runmax[ct] = fmaxf(runmax[ct], __shfl_xor(runmax[ct], 16, 32));

  __shared__ float smax[8][128];
  if (lane < 16) {
    #pragma unroll
    for (int ct = 0; ct < 8; ++ct) smax[wave][ct*16 + lane] = runmax[ct];
  }
  __syncthreads();
  if (tid < 128) {
    float pm = smax[0][tid];
    #pragma unroll
    for (int w = 1; w < 8; ++w) pm = fmaxf(pm, smax[w][tid]);
    pmax[blockIdx.x * 128 + tid] = pm;
  }
}

// ---------------------------------------------------------------------------
// Pass 2: single block — reduce maxes (+scalar tail), loc/env embeds,
// pre-layer, LayerNorm, LSTM step, small softmaxes, att, att@W2, att·b2
// ---------------------------------------------------------------------------
__global__ __launch_bounds__(256) void k_small(
  const float* loc, const float* env,
  const float* U0, const float* U1, const float* U2,
  const float* h0, const float* c0,
  const float* W_loc, const float* b_loc, const float* W_env, const float* b_env,
  const float* W1_0, const float* b1_0, const float* W2_0, const float* b2_0,
  const float* W1_1, const float* b1_1, const float* W2_1, const float* b2_1,
  const float* W1_2, const float* b1_2, const float* W2_2, const float* b2_2,
  const float* W_pre, const float* b_pre, const float* ln_g, const float* ln_b,
  const float* W_ih, const float* W_hh, const float* b_ih, const float* b_hh,
  const float* W_mx, const float* b_mx, const float* W_my, const float* b_my,
  const float* W_enum, const float* b_enum, const float* W_att, const float* b_att,
  const float* pmax, float* aW, float* cconst, float* out, int N, int total)
{
  __shared__ float x640[640];
  __shared__ float xp[128];     // pre-layer out, later reused for att
  __shared__ float xln[128];
  __shared__ float hnew[128];
  __shared__ float gates[512];
  __shared__ float red[2];
  __shared__ float lx[9], ly[9], le[3];
  const int tid = threadIdx.x;

  const float* Us[3]  = {U0,U1,U2};
  const float* W1s[3] = {W1_0,W1_1,W1_2};
  const float* b1s[3] = {b1_0,b1_1,b1_2};
  const float* W2s[3] = {W2_0,W2_1,W2_2};
  const float* b2s[3] = {b2_0,b2_1,b2_2};
  const int tiles = N >> 4, rem = N & 15;

  if (tid < 128) {
    for (int ds = 0; ds < 3; ++ds) {
      float pm = NEG_INF;
      const float* p = pmax + ds * GRIDX * 128 + tid;
      for (int b = 0; b < GRIDX; ++b) pm = fmaxf(pm, p[b * 128]);
      for (int r = 0; r < rem; ++r) {            // scalar tail units
        int m = tiles * 16 + r;
        float u0 = Us[ds][m*3], u1 = Us[ds][m*3+1], u2 = Us[ds][m*3+2];
        float acc = 0.f;
        for (int k = 0; k < 32; ++k) {
          float h = fmaxf(u0*W1s[ds][k*3] + u1*W1s[ds][k*3+1] +
                          u2*W1s[ds][k*3+2] + b1s[ds][k], 0.f);
          acc += h * W2s[ds][tid*32 + k];
        }
        pm = fmaxf(pm, acc);
      }
      x640[256 + ds*128 + tid] = pm + b2s[ds][tid];
    }
    x640[tid]     = fmaxf(loc[0]*W_loc[tid*2] + loc[1]*W_loc[tid*2+1] + b_loc[tid], 0.f);
    x640[128+tid] = fmaxf(env[0]*W_env[tid*2] + env[1]*W_env[tid*2+1] + b_env[tid], 0.f);
  }
  __syncthreads();

  if (tid < 128) {                               // pre-layer 640 -> 128, relu
    float s = b_pre[tid];
    const float* w = W_pre + tid * 640;
    for (int j = 0; j < 640; ++j) s += x640[j] * w[j];
    xp[tid] = fmaxf(s, 0.f);
  }
  __syncthreads();

  if (tid == 0) {                                // LayerNorm stats (serial = deterministic)
    float mu = 0.f;  for (int j = 0; j < 128; ++j) mu += xp[j];
    mu *= (1.f/128.f);
    float var = 0.f; for (int j = 0; j < 128; ++j) { float d = xp[j]-mu; var += d*d; }
    var *= (1.f/128.f);
    red[0] = mu; red[1] = rsqrtf(var + 1e-5f);
  }
  __syncthreads();
  if (tid < 128) xln[tid] = (xp[tid]-red[0])*red[1]*ln_g[tid] + ln_b[tid];
  __syncthreads();

  for (int g = tid; g < 512; g += 256) {         // LSTM gates
    float s = b_ih[g] + b_hh[g];
    const float* wi = W_ih + g * 128; const float* wh = W_hh + g * 128;
    for (int j = 0; j < 128; ++j) s += xln[j]*wi[j] + h0[j]*wh[j];
    gates[g] = s;
  }
  __syncthreads();

  if (tid < 128) {
    float ig = 1.f/(1.f + __expf(-gates[tid]));
    float fg = 1.f/(1.f + __expf(-gates[128+tid]));
    float gg = tanhf(gates[256+tid]);
    float og = 1.f/(1.f + __expf(-gates[384+tid]));
    float cn = fg * c0[tid] + ig * gg;
    float hn = og * tanhf(cn);
    out[21 + total + tid]       = hn;            // h_new
    out[21 + total + 128 + tid] = cn;            // c_new
    hnew[tid] = hn;
  }
  __syncthreads();

  if (tid < 9) {
    float sx = b_mx[tid], sy = b_my[tid];
    for (int j = 0; j < 128; ++j) { sx += hnew[j]*W_mx[tid*128+j]; sy += hnew[j]*W_my[tid*128+j]; }
    lx[tid] = sx; ly[tid] = sy;
  }
  if (tid < 3) {
    float se = b_enum[tid];
    for (int j = 0; j < 128; ++j) se += hnew[j]*W_enum[tid*128+j];
    le[tid] = se;
  }
  __syncthreads();

  if (tid == 0) {                                // tiny softmaxes
    float o9[9];
    {
      float m = lx[0]; for (int i = 1; i < 9; ++i) m = fmaxf(m, lx[i]);
      float s = 0.f;   for (int i = 0; i < 9; ++i) { o9[i] = __expf(lx[i]-m); s += o9[i]; }
      for (int i = 0; i < 9; ++i) out[3+i] = o9[i]/s;
    }
    {
      float m = ly[0]; for (int i = 1; i < 9; ++i) m = fmaxf(m, ly[i]);
      float s = 0.f;   for (int i = 0; i < 9; ++i) { o9[i] = __expf(ly[i]-m); s += o9[i]; }
      for (int i = 0; i < 9; ++i) out[12+i] = o9[i]/s;
    }
    {
      float m  = fmaxf(le[0], fmaxf(le[1], le[2]));
      float e0 = __expf(le[0]-m), e1 = __expf(le[1]-m), e2 = __expf(le[2]-m);
      float s  = e0+e1+e2;
      out[0] = e0/s; out[1] = e1/s; out[2] = e2/s;
    }
  }

  if (tid < 128) {                               // att = hnew @ W_att.T + b_att (reuse xp)
    float s = b_att[tid];
    const float* w = W_att + tid * 128;
    for (int j = 0; j < 128; ++j) s += hnew[j]*w[j];
    xp[tid] = s;
  }
  __syncthreads();

  if (tid < 96) {                                // aW[ds][k] = sum_d att[d]*W2_ds[d][k]
    int ds = tid >> 5, k = tid & 31;
    float s = 0.f;
    for (int d = 0; d < 128; ++d) s += xp[d] * W2s[ds][d*32 + k];
    aW[tid] = s;
  }
  if (tid < 3) {                                 // const = att · b2_ds
    float s = 0.f;
    for (int d = 0; d < 128; ++d) s += xp[d] * b2s[tid][d];
    cconst[tid] = s;
  }
}

// ---------------------------------------------------------------------------
// Pass 3: score_n = aW · relu(u_n@W1.T + b1) + const  (f32) + per-block max
// ---------------------------------------------------------------------------
__global__ __launch_bounds__(256) void k_scores(
    const float* __restrict__ U, const float* __restrict__ W1,
    const float* __restrict__ b1, const float* __restrict__ aW,
    const float* __restrict__ cc, float* __restrict__ scores,
    float* __restrict__ bmax, int N)
{
  __shared__ float sW1[96], sb1[32], saW[32], sred[256];
  const int tid = threadIdx.x;
  if (tid < 96) sW1[tid] = W1[tid];
  if (tid < 32) { sb1[tid] = b1[tid]; saW[tid] = aW[tid]; }
  __syncthreads();

  const int idx = blockIdx.x * 256 + tid;
  float sc = NEG_INF;
  if (idx < N) {
    float u0 = U[idx*3], u1 = U[idx*3+1], u2 = U[idx*3+2];
    float s = cc[0];
    #pragma unroll
    for (int k = 0; k < 32; ++k) {
      float h = fmaxf(u0*sW1[k*3] + u1*sW1[k*3+1] + u2*sW1[k*3+2] + sb1[k], 0.f);
      s += saW[k] * h;
    }
    scores[idx] = s;
    sc = s;
  }
  sred[tid] = sc;
  __syncthreads();
  for (int off = 128; off > 0; off >>= 1) {
    if (tid < off) sred[tid] = fmaxf(sred[tid], sred[tid+off]);
    __syncthreads();
  }
  if (tid == 0) bmax[blockIdx.x] = sred[0];
}

__global__ __launch_bounds__(256) void k_rmax(const float* __restrict__ bmax,
                                              int n, float* __restrict__ gmax)
{
  __shared__ float sr[256];
  const int tid = threadIdx.x;
  float m = NEG_INF;
  for (int i = tid; i < n; i += 256) m = fmaxf(m, bmax[i]);
  sr[tid] = m; __syncthreads();
  for (int off = 128; off > 0; off >>= 1) {
    if (tid < off) sr[tid] = fmaxf(sr[tid], sr[tid+off]);
    __syncthreads();
  }
  if (tid == 0) gmax[0] = sr[0];
}

__global__ __launch_bounds__(256) void k_exp(float* __restrict__ outT,
                                             const float* __restrict__ gmax,
                                             float* __restrict__ bsum, int total)
{
  __shared__ float sr[256];
  const int tid = threadIdx.x;
  const int idx = blockIdx.x * 256 + tid;
  float e = 0.f;
  if (idx < total) { e = __expf(outT[idx] - gmax[0]); outT[idx] = e; }
  sr[tid] = e; __syncthreads();
  for (int off = 128; off > 0; off >>= 1) {
    if (tid < off) sr[tid] += sr[tid+off];
    __syncthreads();
  }
  if (tid == 0) bsum[blockIdx.x] = sr[0];
}

__global__ __launch_bounds__(256) void k_rsum(const float* __restrict__ bsum,
                                              int n, float* __restrict__ inv)
{
  __shared__ float sr[256];
  const int tid = threadIdx.x;
  float s = 0.f;
  for (int i = tid; i < n; i += 256) s += bsum[i];
  sr[tid] = s; __syncthreads();
  for (int off = 128; off > 0; off >>= 1) {
    if (tid < off) sr[tid] += sr[tid+off];
    __syncthreads();
  }
  if (tid == 0) inv[0] = 1.0f / sr[0];
}

__global__ __launch_bounds__(256) void k_scale(float* __restrict__ outT,
                                               const float* __restrict__ inv,
                                               int total)
{
  const int idx = blockIdx.x * 256 + threadIdx.x;
  if (idx < total) outT[idx] *= inv[0];
}

// ---------------------------------------------------------------------------
extern "C" void kernel_launch(void* const* d_in, const int* in_sizes, int n_in,
                              void* d_out, int out_size, void* d_ws, size_t ws_size,
                              hipStream_t stream)
{
  (void)n_in; (void)out_size; (void)ws_size;
  const float* in[39];
  for (int i = 0; i < 39; ++i) in[i] = (const float*)d_in[i];

  const int N     = in_sizes[2] / 3;
  const int total = 3 * N;
  const int NBX   = (N + 255) / 256;
  const int NBT   = (total + 255) / 256;

  float* out = (float*)d_out;
  float* ws  = (float*)d_ws;

  float* pmax   = ws;                       // 3*GRIDX*128
  float* aW     = pmax + 3 * GRIDX * 128;   // 96
  float* cc     = aW + 96;                  // 3
  float* gmax   = cc + 3;                   // 1
  float* inv    = gmax + 1;                 // 1
  float* bmax   = inv + 1;                  // 3*NBX
  float* bsum   = bmax + 3 * NBX;           // NBT
  float* tgt    = out + 21;                 // probs_target region (scratch for scores)

  const float* U [3] = {in[2],  in[3],  in[4]};
  const float* W1[3] = {in[11], in[15], in[19]};
  const float* b1[3] = {in[12], in[16], in[20]};
  const float* W2[3] = {in[13], in[17], in[21]};

  for (int ds = 0; ds < 3; ++ds)
    k_unit_mlp_max<<<GRIDX, 256, 0, stream>>>(U[ds], W1[ds], b1[ds], W2[ds],
                                              pmax + ds * GRIDX * 128, N);

  k_small<<<1, 256, 0, stream>>>(
      in[0], in[1], U[0], U[1], U[2], in[5], in[6],
      in[7], in[8], in[9], in[10],
      in[11], in[12], in[13], in[14],
      in[15], in[16], in[17], in[18],
      in[19], in[20], in[21], in[22],
      in[23], in[24], in[25], in[26],
      in[27], in[28], in[29], in[30],
      in[31], in[32], in[33], in[34], in[35], in[36], in[37], in[38],
      pmax, aW, cc, out, N, total);

  for (int ds = 0; ds < 3; ++ds)
    k_scores<<<NBX, 256, 0, stream>>>(U[ds], W1[ds], b1[ds], aW + ds * 32,
                                      cc + ds, tgt + ds * N, bmax + ds * NBX, N);

  k_rmax <<<1,   256, 0, stream>>>(bmax, 3 * NBX, gmax);
  k_exp  <<<NBT, 256, 0, stream>>>(tgt, gmax, bsum, total);
  k_rsum <<<1,   256, 0, stream>>>(bsum, NBT, inv);
  k_scale<<<NBT, 256, 0, stream>>>(tgt, inv, total);
}
